// DecoderModel_74758200754339
// MI455X (gfx1250) — matrix operand order repaired and checked by reading, not compiled, not run on hardware
//
#include <hip/hip_runtime.h>
#include <hip/hip_bf16.h>

// ---------------------------------------------------------------------------
// Pointer-network LSTM decoder for MI455X (gfx1250).
//
// Strategy (see analysis): persistent 32-block kernel runs the 1024-step scan.
// Weights are pre-converted to bf16 and pre-swizzled into WMMA B-operand tile
// layout so each wave's 16x32 tile load is a fully-coalesced 1KB read.
// Mat-vecs use v_wmma_f32_16x16x32_bf16 with the activation vector broadcast
// into all 16 A-rows (all D rows then hold the mat-vec result).
// Two grid-wide barriers per step (gates->h, logits->argmax).
// ---------------------------------------------------------------------------

typedef __bf16 bf16_t;
typedef __attribute__((ext_vector_type(16))) __bf16 v16bf;
typedef __attribute__((ext_vector_type(8)))  __bf16 v8bf;
typedef __attribute__((ext_vector_type(8)))  float  v8f;

#define SEQ_LEN   1024
#define NODE_FEAT 256
#define HIDDEN    1024
#define GATES4    4096
#define OUTN      1024
#define NBLK      32
#define NTHR      256          // 8 waves/block, 256 waves total

// ---- workspace layout (bytes), all 256-aligned --------------------------
#define OFF_WIH    ((size_t)0)                       // 4096*256  bf16 = 2 MiB
#define OFF_WHH    ((size_t)2097152)                 // 4096*1024 bf16 = 8 MiB
#define OFF_WOUT   ((size_t)10485760)                // 1024*1024 bf16 = 2 MiB
#define OFF_XBF    ((size_t)12582912)                // 1024*256  bf16 = 512 KiB
#define OFF_GATES  ((size_t)13107200)                // 4096 f32
#define OFF_LOGITS ((size_t)13123584)                // 1024 f32
#define OFF_BAR    ((size_t)13127680)                // 2 x u32

union ABf { v16bf v; v8bf h[2]; };

__device__ __forceinline__ v8f wmma_bf16(v16bf a, v16bf b, v8f c) {
  return __builtin_amdgcn_wmma_f32_16x16x32_bf16(false, a, false, b,
                                                 (short)0, c, false, false);
}

// Broadcast activation vector chunk [kbase, kbase+32) into the 16x32 A tile:
// all 16 rows identical.  Per the CDNA5 bf16 A layout, lanes 0-15 hold
// K 0..7 / 16..23 and lanes 16-31 hold K 8..15 / 24..31 of the chunk.
__device__ __forceinline__ v16bf load_a_bcast(const bf16_t* vec, int kbase,
                                              int half16) {
  ABf a;
  a.h[0] = *(const v8bf*)(vec + kbase + 8 * half16);
  a.h[1] = *(const v8bf*)(vec + kbase + 16 + 8 * half16);
  return a.v;
}

__device__ __forceinline__ float sigmoidf_(float x) {
  return 1.0f / (1.0f + __expf(-x));
}

// ---------------------------------------------------------------------------
// Setup kernels (run every launch; deterministic).
// ---------------------------------------------------------------------------

// Re-swizzle a row-major f32 weight matrix (R outputs x K inputs) into bf16
// WMMA B-operand tiles:  packed[((wblk*(K/32)+kc)*32 + lane)*16 + e]
//   = W[wblk*16 + lane%16, kc*32 + 8*(lane/16) + e%8 + 16*(e/8)]
__global__ void pack_b_tiles(const float* __restrict__ src,
                             bf16_t* __restrict__ dst, int K, int total) {
  int idx = blockIdx.x * blockDim.x + threadIdx.x;
  if (idx >= total) return;
  int e    = idx & 15;
  int l    = (idx >> 4) & 31;
  int kc   = (idx >> 9) % (K >> 5);
  int wblk = idx / ((K >> 5) << 9);
  int row  = (wblk << 4) + (l & 15);
  int col  = (kc << 5) + ((l >> 4) << 3) + (e & 7) + ((e >> 3) << 4);
  dst[idx] = (bf16_t)src[row * K + col];
}

__global__ void cvt_to_bf16(const float* __restrict__ src,
                            bf16_t* __restrict__ dst, int n) {
  int i = blockIdx.x * blockDim.x + threadIdx.x;
  if (i < n) dst[i] = (bf16_t)src[i];
}

__global__ void init_barrier(unsigned* bar) {
  if (threadIdx.x == 0) { bar[0] = 0u; bar[1] = 0u; }
}

// ---------------------------------------------------------------------------
// Grid-wide barrier (all NBLK blocks resident).
// ---------------------------------------------------------------------------
__device__ __forceinline__ void grid_sync(unsigned* arrive,
                                          volatile unsigned* release) {
  __threadfence();
  __syncthreads();
  if (threadIdx.x == 0) {
    unsigned g = *release;
    if (atomicAdd(arrive, 1u) == (unsigned)(NBLK - 1)) {
      *arrive = 0u;
      __threadfence();
      *release = g + 1u;
    } else {
      while (*release == g) { __builtin_amdgcn_s_sleep(2); }
    }
  }
  __syncthreads();
  __threadfence();
}

// ---------------------------------------------------------------------------
// Main persistent decode kernel.
// ---------------------------------------------------------------------------
__global__ __launch_bounds__(NTHR) void decoder_scan(
    const float* __restrict__ enc,        // (1024,1024); row 1023 is c0
    const float* __restrict__ b_ih,       // (4096,)
    const float* __restrict__ b_hh,       // (4096,)
    const float* __restrict__ b_out,      // (1024,)
    const v16bf* __restrict__ wih,        // packed tiles, K=256
    const v16bf* __restrict__ whh,        // packed tiles, K=1024
    const v16bf* __restrict__ wout,       // packed tiles, K=1024
    const bf16_t* __restrict__ xbf,       // (1024,256) bf16
    float* __restrict__ gates,            // (4096,)
    float* __restrict__ logits,           // (1024,)
    unsigned* __restrict__ bar,           // [arrive, release]
    float* __restrict__ out_idx,          // (1024,)
    float* __restrict__ out_logp) {       // (1024,)
  __shared__ __align__(32) bf16_t lds_h[HIDDEN];
  __shared__ float sh_visited[OUTN];      // used by block 0 only
  __shared__ float red_f[NTHR];
  __shared__ float red_m[NTHR];
  __shared__ int   red_i[NTHR];

  const int tid    = threadIdx.x;
  const int lane   = tid & 31;
  const int wave   = tid >> 5;                 // 0..7
  const int gw     = blockIdx.x * 8 + wave;    // 0..255 global wave id
  const int half16 = lane >> 4;
  const int col    = (gw << 4) + (lane & 15);  // this wave's output columns

  const float* c0 = enc + (size_t)(SEQ_LEN - 1) * HIDDEN;
  unsigned* arrive = bar;
  volatile unsigned* release = bar + 1;

  for (int j = tid; j < HIDDEN; j += NTHR) {
    lds_h[j] = (bf16_t)0.0f;
    sh_visited[j] = 0.0f;
  }
  __syncthreads();

  for (int t = 0; t < SEQ_LEN; ++t) {
    // ---- phase 1: gates[4096] = x_t W_ih^T + h W_hh^T + b_ih + b_hh -----
    {
      float bsum = b_ih[col] + b_hh[col];
      v8f acc;
      #pragma unroll
      for (int i = 0; i < 8; ++i) acc[i] = bsum;

      const bf16_t* xrow = xbf + (size_t)t * NODE_FEAT;
      #pragma unroll
      for (int kc = 0; kc < NODE_FEAT / 32; ++kc) {
        v16bf a = load_a_bcast(xrow, kc * 32, half16);
        v16bf b = wih[(gw * (NODE_FEAT / 32) + kc) * 32 + lane];
        acc = wmma_bf16(a, b, acc);
      }
      for (int kc = 0; kc < HIDDEN / 32; ++kc) {
        v16bf a = load_a_bcast(lds_h, kc * 32, half16);
        v16bf b = whh[(gw * (HIDDEN / 32) + kc) * 32 + lane];
        acc = wmma_bf16(a, b, acc);
      }
      if (lane < 16) gates[col] = acc[0];   // all D rows identical; row 0
    }
    grid_sync(arrive, release);

    // ---- phase 2: every block computes the full h (redundant, no sync) --
    for (int j = tid; j < HIDDEN; j += NTHR) {
      float ig = sigmoidf_(gates[j]);
      float fg = sigmoidf_(gates[j + HIDDEN]);
      float gg = tanhf(gates[j + 2 * HIDDEN]);
      float og = sigmoidf_(gates[j + 3 * HIDDEN]);
      float c  = fg * c0[j] + ig * gg;      // quirk: c0 fixed every step
      lds_h[j] = (bf16_t)(og * tanhf(c));
    }
    __syncthreads();

    // ---- phase 3: logits[1024] = h W_out^T + b_out (waves 0..63) --------
    if (gw < OUTN / 16) {
      float bo = b_out[col];
      v8f acc;
      #pragma unroll
      for (int i = 0; i < 8; ++i) acc[i] = bo;
      for (int kc = 0; kc < HIDDEN / 32; ++kc) {
        v16bf a = load_a_bcast(lds_h, kc * 32, half16);
        v16bf b = wout[(gw * (HIDDEN / 32) + kc) * 32 + lane];
        acc = wmma_bf16(a, b, acc);
      }
      if (lane < 16) logits[col] = acc[0];
    }
    grid_sync(arrive, release);

    // ---- phase 4: block 0 does softmax + masked argmax ------------------
    // masked = softmax(logits) - visited; since probs > 0 and visited terms
    // drop below 0, argmax(masked) = argmax over unvisited logits and
    // log(max(masked)) = l* - max(logits) - log(sum exp(l - max)).
    if (blockIdx.x == 0) {
      float m = -3.4e38f;
      for (int j = tid; j < OUTN; j += NTHR) m = fmaxf(m, logits[j]);
      red_f[tid] = m;
      __syncthreads();
      for (int s = NTHR / 2; s > 0; s >>= 1) {
        if (tid < s) red_f[tid] = fmaxf(red_f[tid], red_f[tid + s]);
        __syncthreads();
      }
      float maxall = red_f[0];
      __syncthreads();

      float se = 0.0f, bv = -3.4e38f;
      int bi = 0x7fffffff;
      for (int j = tid; j < OUTN; j += NTHR) {
        float l = logits[j];
        se += __expf(l - maxall);
        if (sh_visited[j] == 0.0f && (l > bv || (l == bv && j < bi))) {
          bv = l; bi = j;
        }
      }
      red_f[tid] = se; red_m[tid] = bv; red_i[tid] = bi;
      __syncthreads();
      for (int s = NTHR / 2; s > 0; s >>= 1) {
        if (tid < s) {
          red_f[tid] += red_f[tid + s];
          float mv = red_m[tid + s]; int mi = red_i[tid + s];
          if (mv > red_m[tid] || (mv == red_m[tid] && mi < red_i[tid])) {
            red_m[tid] = mv; red_i[tid] = mi;
          }
        }
        __syncthreads();
      }
      if (tid == 0) {
        int ptr = red_i[0];
        out_idx[t]  = (float)ptr;
        out_logp[t] = red_m[0] - maxall - __logf(red_f[0]);
        sh_visited[ptr] = 1.0f;
      }
      __syncthreads();
    }
    // no barrier needed here: block 0 can't pass next step's phase-1 barrier
    // until phase 4 is done, so logits can't be overwritten early.
  }
}

// ---------------------------------------------------------------------------
extern "C" void kernel_launch(void* const* d_in, const int* in_sizes, int n_in,
                              void* d_out, int out_size, void* d_ws,
                              size_t ws_size, hipStream_t stream) {
  const float* input   = (const float*)d_in[0];  // (1024,256)
  const float* enc     = (const float*)d_in[1];  // (1024,1024)
  const float* W_ih    = (const float*)d_in[2];  // (4096,256)
  const float* W_hh    = (const float*)d_in[3];  // (4096,1024)
  const float* b_ih    = (const float*)d_in[4];
  const float* b_hh    = (const float*)d_in[5];
  const float* W_out   = (const float*)d_in[6];  // (1024,1024)
  const float* b_out   = (const float*)d_in[7];

  char* ws = (char*)d_ws;
  bf16_t* wih_p  = (bf16_t*)(ws + OFF_WIH);
  bf16_t* whh_p  = (bf16_t*)(ws + OFF_WHH);
  bf16_t* wout_p = (bf16_t*)(ws + OFF_WOUT);
  bf16_t* xbf    = (bf16_t*)(ws + OFF_XBF);
  float*  gates  = (float*)(ws + OFF_GATES);
  float*  logits = (float*)(ws + OFF_LOGITS);
  unsigned* bar  = (unsigned*)(ws + OFF_BAR);

  float* out_idx  = (float*)d_out;
  float* out_logp = (float*)d_out + SEQ_LEN;

  // Setup: bf16 conversion + B-tile swizzle + barrier reset (every launch).
  {
    int tot = GATES4 * NODE_FEAT;
    pack_b_tiles<<<(tot + 255) / 256, 256, 0, stream>>>(W_ih, wih_p, NODE_FEAT, tot);
  }
  {
    int tot = GATES4 * HIDDEN;
    pack_b_tiles<<<(tot + 255) / 256, 256, 0, stream>>>(W_hh, whh_p, HIDDEN, tot);
  }
  {
    int tot = OUTN * HIDDEN;
    pack_b_tiles<<<(tot + 255) / 256, 256, 0, stream>>>(W_out, wout_p, HIDDEN, tot);
  }
  {
    int tot = SEQ_LEN * NODE_FEAT;
    cvt_to_bf16<<<(tot + 255) / 256, 256, 0, stream>>>(input, xbf, tot);
  }
  init_barrier<<<1, 64, 0, stream>>>(bar);

  decoder_scan<<<NBLK, NTHR, 0, stream>>>(enc, b_ih, b_hh, b_out,
                                          (const v16bf*)wih_p,
                                          (const v16bf*)whh_p,
                                          (const v16bf*)wout_p,
                                          xbf, gates, logits, bar,
                                          out_idx, out_logp);
}
